// Encoder_23682449670436
// MI455X (gfx1250) — matrix operand order repaired and checked by reading, not compiled, not run on hardware
//
#include <hip/hip_runtime.h>
#include <math.h>
#include <stdint.h>

// ---------------------------------------------------------------------------
// Problem constants (from reference)
// ---------------------------------------------------------------------------
#define N_CODES 4880
#define T_DIM   4096
#define VDIM    512
#define HDIM    512
#define G3      1536   // 3*HDIM

typedef __bf16 bf16_t;
typedef bf16_t bf16x16 __attribute__((ext_vector_type(16)));
typedef float  f32x8   __attribute__((ext_vector_type(8)));
typedef int    v4i     __attribute__((vector_size(4 * sizeof(int))));

// GEMM tiling: 256 threads = 8 waves; block tile 128(M) x 64(N) x 32(K)
#define BM  128
#define BN  64
#define BK  32
#define LDK 40   // padded K-stride (bf16): 80B rows -> every 16B chunk aligned

// ---------------------------------------------------------------------------
// CDNA5 async global->LDS path (ASYNCcnt), guarded so the file always compiles
// Builtin prototype (from compiler diagnostic): (v4i32 AS1*, v4i32 AS3*, imm, imm)
// ---------------------------------------------------------------------------
#define AS1_V4I(p) ((__attribute__((address_space(1))) v4i*)(uintptr_t)(p))
#define AS3_V4I(p) ((__attribute__((address_space(3))) v4i*)(uint32_t)(uintptr_t)(p))

#if defined(__AMDGCN__) && __has_builtin(__builtin_amdgcn_global_load_async_to_lds_b128)
  #define HAVE_ASYNC_LDS 1
  #define ASYNC_CP_B128(gsrc, ldst) \
      __builtin_amdgcn_global_load_async_to_lds_b128(AS1_V4I(gsrc), AS3_V4I(ldst), 0, 0)
  #if __has_builtin(__builtin_amdgcn_s_wait_asynccnt)
    #define WAIT_ASYNC(n) __builtin_amdgcn_s_wait_asynccnt(n)
  #else
    #define WAIT_ASYNC(n) asm volatile("s_wait_asynccnt %0" :: "i"(n) : "memory")
  #endif
#else
  #define HAVE_ASYNC_LDS 0
  #define WAIT_ASYNC(n) ((void)0)
#endif

// ---------------------------------------------------------------------------
// GEMM 1: Vbf[4096,512] = H^T (4096x4880) * X (4880x512)
//   A[m,k] = H[k, m]   (H row-major (4880, 4096))
//   f32 inputs -> bf16 in LDS -> v_wmma_f32_16x16x32_bf16 -> bf16 out
//   (f32->bf16 conversion forces the VGPR staging path here.)
// ---------------------------------------------------------------------------
__global__ __launch_bounds__(256) void gemm_visit_kernel(
    const float* __restrict__ H, const float* __restrict__ X,
    bf16_t* __restrict__ Vbf)
{
    __shared__ bf16_t Asb[BM * LDK];   // [m][k]
    __shared__ bf16_t Bsb[BN * LDK];   // [n][k]  (B^T layout)

    const int tid   = threadIdx.x;
    const int wave  = tid >> 5;
    const int lane  = tid & 31;
    const int lh    = lane >> 4;       // lane half (0: lanes 0-15, 1: 16-31)
    const int lm    = lane & 15;
    const int mBase = blockIdx.x * BM;
    const int nBase = blockIdx.y * BN;

    f32x8 acc[4] = {};

    for (int k0 = 0; k0 < N_CODES; k0 += BK) {
        if (k0 + BK < N_CODES) {
            __builtin_prefetch(&H[(size_t)(k0 + BK) * T_DIM + mBase + tid], 0, 1);
            __builtin_prefetch(&X[(size_t)(k0 + BK) * VDIM + nBase], 0, 1);
        }
        // stage A tile: 128m x 32k, read H row k contiguously in m (coalesced)
        for (int idx = tid; idx < BM * BK; idx += 256) {
            int m  = idx & (BM - 1);
            int k  = idx >> 7;
            int gk = k0 + k;
            float v = (gk < N_CODES) ? H[(size_t)gk * T_DIM + mBase + m] : 0.0f;
            Asb[m * LDK + k] = (bf16_t)v;
        }
        // stage B tile: 32k x 64n, read X row k contiguously in n (coalesced)
        for (int idx = tid; idx < BN * BK; idx += 256) {
            int n  = idx & (BN - 1);
            int k  = idx >> 6;
            int gk = k0 + k;
            float v = (gk < N_CODES) ? X[(size_t)gk * VDIM + nBase + n] : 0.0f;
            Bsb[n * LDK + k] = (bf16_t)v;
        }
        __syncthreads();

        // A fragment: 16x32 bf16 per wave (ISA 7.12.2 16-bit A layout)
        bf16x16 a;
        const int am = wave * 16 + lm;
        #pragma unroll
        for (int e = 0; e < 16; ++e) {
            int v = e >> 1;
            int k = ((e < 8) ? 0 : 16) + (lh << 3) + ((v & 3) << 1) + (e & 1);
            a[e] = Asb[am * LDK + k];
        }
        #pragma unroll
        for (int nt = 0; nt < 4; ++nt) {
            bf16x16 b;
            const int bn = nt * 16 + lm;
            #pragma unroll
            for (int e = 0; e < 16; ++e) {
                int k = ((e < 8) ? 0 : 16) + (lh << 3) + (e & 7);
                b[e] = Bsb[bn * LDK + k];
            }
            acc[nt] = __builtin_amdgcn_wmma_f32_16x16x32_bf16(
                false, a, false, b, (short)0, acc[nt], false, false);
        }
        __syncthreads();
    }

    // epilogue: C layout -> VGPR r holds (M = r + 8*lh, N = lane&15)
    const int rowB = mBase + wave * 16 + (lh << 3);
    #pragma unroll
    for (int nt = 0; nt < 4; ++nt) {
        const int col = nBase + nt * 16 + lm;
        #pragma unroll
        for (int r = 0; r < 8; ++r) {
            Vbf[(size_t)(rowB + r) * VDIM + col] = (bf16_t)acc[nt][r];
        }
    }
}

// ---------------------------------------------------------------------------
// GEMM 2: gi[4096,1536] = Vbf (4096x512 bf16) * W_ih^T + b_ih
//   B[k,n] = W_ih[n, k]   (W_ih row-major (1536, 512))
//   A operand is raw bf16 -> staged with GLOBAL_LOAD_ASYNC_TO_LDS_B128,
//   double-buffered LDS, synchronized with s_wait_asynccnt.
// ---------------------------------------------------------------------------
__global__ __launch_bounds__(256) void gemm_gi_kernel(
    const bf16_t* __restrict__ V, const float* __restrict__ W_ih,
    const float* __restrict__ b_ih, float* __restrict__ gi)
{
    __shared__ bf16_t Asb[2][BM * LDK];
    __shared__ bf16_t Bsb[2][BN * LDK];

    const int tid   = threadIdx.x;
    const int wave  = tid >> 5;
    const int lane  = tid & 31;
    const int lh    = lane >> 4;
    const int lm    = lane & 15;
    const int mBase = blockIdx.x * BM;
    const int nBase = blockIdx.y * BN;

    const int NKC = VDIM / BK;   // 16 K-chunks, no tail

    // --- staging helpers -------------------------------------------------
    auto stageA = [&](int kc, int buf) {
#if HAVE_ASYNC_LDS
        // A tile = 128 rows x 64B; 512 x 16B chunks; 2 per thread.
        // LDS row pitch = LDK*2 = 80B (multiple of 16) -> b128 stays aligned.
        #pragma unroll
        for (int c = 0; c < 2; ++c) {
            int chunk = tid + c * 256;
            int row   = chunk >> 2;
            int part  = chunk & 3;
            const bf16_t* src = V + (size_t)(mBase + row) * VDIM + kc * BK + part * 8;
            bf16_t*       dst = &Asb[buf][row * LDK + part * 8];
            ASYNC_CP_B128(src, dst);
        }
#else
        for (int idx = tid; idx < BM * BK; idx += 256) {
            int k = idx & 31;
            int m = idx >> 5;
            Asb[buf][m * LDK + k] = V[(size_t)(mBase + m) * VDIM + kc * BK + k];
        }
#endif
    };
    auto stageB = [&](int kc, int buf) {
        // W_ih[n, k] f32 -> bf16; k fastest (coalesced b32 loads)
        for (int idx = tid; idx < BN * BK; idx += 256) {
            int k = idx & 31;
            int n = idx >> 5;
            Bsb[buf][n * LDK + k] =
                (bf16_t)W_ih[(size_t)(nBase + n) * VDIM + kc * BK + k];
        }
    };

    f32x8 acc[4] = {};

    // prologue: stage chunk 0 into buffer 0
    stageB(0, 0);
    stageA(0, 0);

    for (int kc = 0; kc < NKC; ++kc) {
        const int cur = kc & 1;
        if (kc + 1 < NKC) {
            // stage next chunk into the other buffer while current computes
            stageB(kc + 1, 1 - cur);
            stageA(kc + 1, 1 - cur);
            WAIT_ASYNC(2);   // in-order: tile-`kc` async copies retired
        } else {
            WAIT_ASYNC(0);
        }
        __syncthreads();     // B ds_stores + all waves' async data visible

        bf16x16 a;
        const int am = wave * 16 + lm;
        #pragma unroll
        for (int e = 0; e < 16; ++e) {
            int v = e >> 1;
            int k = ((e < 8) ? 0 : 16) + (lh << 3) + ((v & 3) << 1) + (e & 1);
            a[e] = Asb[cur][am * LDK + k];
        }
        #pragma unroll
        for (int nt = 0; nt < 4; ++nt) {
            bf16x16 b;
            const int bn = nt * 16 + lm;
            #pragma unroll
            for (int e = 0; e < 16; ++e) {
                int k = ((e < 8) ? 0 : 16) + (lh << 3) + (e & 7);
                b[e] = Bsb[cur][bn * LDK + k];
            }
            acc[nt] = __builtin_amdgcn_wmma_f32_16x16x32_bf16(
                false, a, false, b, (short)0, acc[nt], false, false);
        }
        __syncthreads();     // readers done before this buffer is re-staged
    }

    const int rowB = mBase + wave * 16 + (lh << 3);
    #pragma unroll
    for (int nt = 0; nt < 4; ++nt) {
        const int col = nBase + nt * 16 + lm;
        const float bias = b_ih[col];
        #pragma unroll
        for (int r = 0; r < 8; ++r) {
            gi[(size_t)(rowB + r) * G3 + col] = acc[nt][r] + bias;
        }
    }
}

// ---------------------------------------------------------------------------
// GRU scan: single workgroup, 512 threads; h in LDS (broadcast reads);
// W_hh (3 MB) stays L2-resident.  Thread j owns rows j / 512+j / 1024+j.
// ---------------------------------------------------------------------------
__global__ __launch_bounds__(512) void gru_scan_kernel(
    const float* __restrict__ gi, const float* __restrict__ W_hh,
    const float* __restrict__ b_hh, float* __restrict__ hs)
{
    __shared__ float h[HDIM];
    const int j = threadIdx.x;
    h[j] = 0.0f;
    __syncthreads();

    const float4* wr = (const float4*)(W_hh + (size_t)j * HDIM);
    const float4* wz = (const float4*)(W_hh + (size_t)(HDIM + j) * HDIM);
    const float4* wn = (const float4*)(W_hh + (size_t)(2 * HDIM + j) * HDIM);
    const float br = b_hh[j];
    const float bz = b_hh[HDIM + j];
    const float bn = b_hh[2 * HDIM + j];

    for (int t = 0; t < T_DIM; ++t) {
        float ar = 0.0f, az = 0.0f, an = 0.0f;
        const float4* h4 = (const float4*)h;
        #pragma unroll 4
        for (int k = 0; k < HDIM / 4; ++k) {
            float4 hv = h4[k];
            float4 a = wr[k];
            float4 b = wz[k];
            float4 c = wn[k];
            ar += a.x * hv.x + a.y * hv.y + a.z * hv.z + a.w * hv.w;
            az += b.x * hv.x + b.y * hv.y + b.z * hv.z + b.w * hv.w;
            an += c.x * hv.x + c.y * hv.y + c.z * hv.z + c.w * hv.w;
        }
        const float* git = gi + (size_t)t * G3;
        float r = 1.0f / (1.0f + __expf(-(git[j] + ar + br)));
        float z = 1.0f / (1.0f + __expf(-(git[HDIM + j] + az + bz)));
        float n = tanhf(git[2 * HDIM + j] + r * (an + bn));
        float hold = h[j];
        float hnew = (1.0f - z) * n + z * hold;
        __syncthreads();
        h[j] = hnew;
        hs[(size_t)t * HDIM + j] = hnew;
        __syncthreads();
    }
}

// ---------------------------------------------------------------------------
// logits[t] = dot(hs[t,:], w_att)
// ---------------------------------------------------------------------------
__global__ __launch_bounds__(256) void logits_kernel(
    const float* __restrict__ hs, const float* __restrict__ w_att,
    float* __restrict__ logits)
{
    const int t = blockIdx.x * blockDim.x + threadIdx.x;
    if (t < T_DIM) {
        const float* row = hs + (size_t)t * HDIM;
        float s = 0.0f;
        #pragma unroll 4
        for (int k = 0; k < HDIM; ++k) s += row[k] * w_att[k];
        logits[t] = s;
    }
}

// ---------------------------------------------------------------------------
// softmax over 4096 logits, in place (single 1024-thread block)
// ---------------------------------------------------------------------------
__global__ __launch_bounds__(1024) void softmax_kernel(float* __restrict__ logits)
{
    __shared__ float red[1024];
    const int tid = threadIdx.x;

    float m = -INFINITY;
    for (int t = tid; t < T_DIM; t += 1024) m = fmaxf(m, logits[t]);
    red[tid] = m;
    __syncthreads();
    for (int s = 512; s > 0; s >>= 1) {
        if (tid < s) red[tid] = fmaxf(red[tid], red[tid + s]);
        __syncthreads();
    }
    const float mx = red[0];
    __syncthreads();

    float sum = 0.0f;
    for (int t = tid; t < T_DIM; t += 1024) sum += __expf(logits[t] - mx);
    red[tid] = sum;
    __syncthreads();
    for (int s = 512; s > 0; s >>= 1) {
        if (tid < s) red[tid] += red[tid + s];
        __syncthreads();
    }
    const float inv = 1.0f / red[0];
    __syncthreads();

    for (int t = tid; t < T_DIM; t += 1024) logits[t] = __expf(logits[t] - mx) * inv;
}

// ---------------------------------------------------------------------------
// out[j] = sum_t alpha[t] * hs[t, j]   (coalesced over j)
// ---------------------------------------------------------------------------
__global__ __launch_bounds__(256) void attend_kernel(
    const float* __restrict__ hs, const float* __restrict__ alpha,
    float* __restrict__ out)
{
    const int j = blockIdx.x * blockDim.x + threadIdx.x;
    if (j < HDIM) {
        float s = 0.0f;
        for (int t = 0; t < T_DIM; ++t) s += alpha[t] * hs[(size_t)t * HDIM + j];
        out[j] = s;
    }
}

// ---------------------------------------------------------------------------
// Host launcher
// ---------------------------------------------------------------------------
extern "C" void kernel_launch(void* const* d_in, const int* in_sizes, int n_in,
                              void* d_out, int out_size, void* d_ws, size_t ws_size,
                              hipStream_t stream)
{
    const float* H     = (const float*)d_in[0];   // (4880, 4096)
    // d_in[1] = TE (unused by reference)
    const float* X     = (const float*)d_in[2];   // (4880, 512)
    const float* W_ih  = (const float*)d_in[3];   // (1536, 512)
    const float* W_hh  = (const float*)d_in[4];   // (1536, 512)
    const float* b_ih  = (const float*)d_in[5];   // (1536,)
    const float* b_hh  = (const float*)d_in[6];   // (1536,)
    const float* w_att = (const float*)d_in[7];   // (1, 512)
    float* out = (float*)d_out;                   // (512,)

    char* ws = (char*)d_ws;
    bf16_t* Vbf   = (bf16_t*)ws;                          //  4 MiB: 4096x512 bf16
    float*  gi    = (float*)(ws + ((size_t)4  << 20));    // 24 MiB: 4096x1536 f32
    float*  hs    = (float*)(ws + ((size_t)28 << 20));    //  8 MiB: 4096x512 f32
    float*  logit = (float*)(ws + ((size_t)36 << 20));    // 16 KiB: 4096 f32

    dim3 blk(256);
    dim3 g1(T_DIM / BM, VDIM / BN);   // 32 x 8
    gemm_visit_kernel<<<g1, blk, 0, stream>>>(H, X, Vbf);

    dim3 g2(T_DIM / BM, G3 / BN);     // 32 x 24
    gemm_gi_kernel<<<g2, blk, 0, stream>>>(Vbf, W_ih, b_ih, gi);

    gru_scan_kernel<<<1, 512, 0, stream>>>(gi, W_hh, b_hh, hs);

    logits_kernel<<<T_DIM / 256, 256, 0, stream>>>(hs, w_att, logit);
    softmax_kernel<<<1, 1024, 0, stream>>>(logit);
    attend_kernel<<<HDIM / 256, 256, 0, stream>>>(hs, logit, out);
}
